// Mamba_Block_1812476199132
// MI455X (gfx1250) — compile-verified, matrix operand order checked by reading, not run on hardware
//
#include <hip/hip_runtime.h>
#include <math.h>

// Mamba block, MI455X (gfx1250, wave32, WMMA bf16 matrix path).
// All GEMMs consume pre-converted bf16 operands (half the HBM bytes),
// fp32 accumulate; scan/elementwise stay fp32.
// GEMM K-loop: ping-pong double buffering (2x unrolled), fragments refilled
// in place -> no register-rotation moves, partial s_wait_loadcnt overlap.

#define B_  4
#define L_  2048
#define DM_ 1024
#define DI_ 2048
#define DS_ 16
#define DR_ 64
#define KC_ 4
#define XD_ 96   // DR + 2*DS

typedef __bf16 bf16_t;
typedef bf16_t v16bf __attribute__((ext_vector_type(16)));
typedef bf16_t v4bf  __attribute__((ext_vector_type(4)));
typedef float  v8f   __attribute__((ext_vector_type(8)));

__device__ __forceinline__ float sigmoidf_(float x) {
  return 1.0f / (1.0f + __expf(-x));
}

template<int NT>
__device__ __forceinline__ void load_frags(const bf16_t* __restrict__ ap,
                                           const bf16_t* __restrict__ bp0,
                                           size_t bstride, int k,
                                           v16bf& a, v16bf (&b)[NT])
{
  a = *(const v16bf*)(ap + k);
#pragma unroll
  for (int t = 0; t < NT; ++t)
    b[t] = *(const v16bf*)(bp0 + (size_t)t * bstride + k);
}

template<int NT>
__device__ __forceinline__ void mma_step(const v16bf& a, const v16bf (&b)[NT],
                                         v8f (&acc)[NT])
{
#pragma unroll
  for (int t = 0; t < NT; ++t)
    acc[t] = __builtin_amdgcn_wmma_f32_16x16x32_bf16(
        false, a, false, b[t], (short)0, acc[t], false, false);
}

// ---------------------------------------------------------------------------
// Generic NT GEMM on bf16 operands: C[m,n] = sum_k A[m,k] * W[n,k]
// A (M x K) row-major bf16, W (N x K) row-major bf16, C row-major f32.
// One wave32 computes NT adjacent 16x16 C tiles along N (shared A fragment,
// NT independent accumulator chains), 8 waves/block.  K % 64 == 0.
// EPI: 0 = none, 1 = softplus(x + bias[n])
// ---------------------------------------------------------------------------
template<int EPI, int NT>
__global__ void __launch_bounds__(256)
wmma_nt_gemm(const bf16_t* __restrict__ A, int lda,
             const bf16_t* __restrict__ W, int ldb,
             float* __restrict__ C, int ldc,
             int N, int K,
             const float* __restrict__ bias)
{
  const int lane   = threadIdx.x & 31;
  const int wave   = threadIdx.x >> 5;
  const int nt0    = (blockIdx.x * 8 + wave) * NT;  // first n-tile of this wave
  const int m_tile = blockIdx.y;
  if (nt0 * 16 >= N) return;                // wave-uniform: EXEC stays all-1s

  // A-fragment, contiguous k order (not the ISA table; B matches): lane L holds A[m = L&15, k = kh..kh+15],
  // kh = (L>>4)*16  -> 16 contiguous bf16 (32 bytes, two b128 loads).
  // B-fragment symmetric: W[n = L&15, k = kh..kh+15].
  const int mrow = m_tile * 16 + (lane & 15);
  const int kh   = (lane >> 4) * 16;

  const bf16_t* ap  = A + (size_t)mrow * lda + kh;
  const bf16_t* bp0 = W + (size_t)(nt0 * 16 + (lane & 15)) * ldb + kh;
  const size_t  bstride = (size_t)16 * ldb;

  v8f acc[NT];
#pragma unroll
  for (int t = 0; t < NT; ++t) acc[t] = (v8f){};

  // ping-pong fragment buffers, refilled in place (no rotation copies)
  v16bf a0, a1;
  v16bf b0[NT], b1[NT];
  load_frags<NT>(ap, bp0, bstride, 0, a0, b0);
  load_frags<NT>(ap, bp0, bstride, 32, a1, b1);

  for (int k0 = 64; k0 < K; k0 += 64) {
    __builtin_prefetch(ap + k0 + 512, 0, 3);
    __builtin_prefetch(bp0 + k0 + 512, 0, 3);
    mma_step<NT>(a0, b0, acc);                        // consume buffer 0
    load_frags<NT>(ap, bp0, bstride, k0, a0, b0);     // refill buffer 0
    mma_step<NT>(a1, b1, acc);                        // consume buffer 1
    load_frags<NT>(ap, bp0, bstride, k0 + 32, a1, b1);// refill buffer 1
  }
  mma_step<NT>(a0, b0, acc);
  mma_step<NT>(a1, b1, acc);

  // C layout: VGPR v, lanes 0-15 -> M = v, lanes 16-31 -> M = v+8; N = lane&15.
  const int rbase = m_tile * 16 + (lane >> 4) * 8;
#pragma unroll
  for (int t = 0; t < NT; ++t) {
    const int ntile = nt0 + t;
    if (ntile * 16 >= N) continue;
    const int cidx = ntile * 16 + (lane & 15);
    float bv_ = (EPI == 1) ? bias[cidx] : 0.0f;
#pragma unroll
    for (int v = 0; v < 8; ++v) {
      float val = acc[t][v];
      if (EPI == 1) {
        float x = val + bv_;
        val = (x > 20.0f) ? x : log1pf(__expf(x));   // softplus
      }
      C[(size_t)(rbase + v) * ldc + cidx] = val;
    }
  }
}

// ---------------------------------------------------------------------------
// Contiguous f32 -> bf16 conversion, 4 elements/thread. n % 1024 == 0.
// ---------------------------------------------------------------------------
__global__ void __launch_bounds__(256)
cvt_f32_bf16(const float* __restrict__ src, bf16_t* __restrict__ dst)
{
  size_t i = ((size_t)blockIdx.x * 256 + threadIdx.x) * 4;
  float4 v = *(const float4*)(src + i);
  v4bf o;
  o[0] = (bf16_t)v.x; o[1] = (bf16_t)v.y; o[2] = (bf16_t)v.z; o[3] = (bf16_t)v.w;
  *(v4bf*)(dst + i) = o;
}

// ---------------------------------------------------------------------------
// Extract dt columns (first 64 of 96) from xdbl and convert to packed bf16.
// One thread per 4 columns.  rows*16 threads total.
// ---------------------------------------------------------------------------
__global__ void __launch_bounds__(256)
extract_dt_bf16(const float* __restrict__ xdbl, bf16_t* __restrict__ dtb)
{
  size_t idx = (size_t)blockIdx.x * 256 + threadIdx.x;   // rows * 16
  size_t r = idx >> 4;
  int    c = (int)(idx & 15) * 4;
  float4 v = *(const float4*)(xdbl + r * XD_ + c);
  v4bf o;
  o[0] = (bf16_t)v.x; o[1] = (bf16_t)v.y; o[2] = (bf16_t)v.z; o[3] = (bf16_t)v.w;
  *(v4bf*)(dtb + r * DR_ + c) = o;
}

// ---------------------------------------------------------------------------
// Causal depthwise conv1d (K=4) + SiLU.  x = xz[:, :, 0:DI] (row stride 2*DI).
// Writes u in f32 (scan precision) and bf16 (x_proj GEMM operand).
// ---------------------------------------------------------------------------
__global__ void __launch_bounds__(256)
conv_silu_kernel(const float* __restrict__ xz, const float* __restrict__ conv_w,
                 const float* __restrict__ conv_b,
                 float* __restrict__ u, bf16_t* __restrict__ ub)
{
  size_t i   = (size_t)blockIdx.x * 256 + threadIdx.x;   // over B*L*DI
  int    d   = (int)(i % DI_);
  size_t row = i / DI_;                                   // b*L + t
  int    t   = (int)(row % L_);
  size_t b0  = row - t;                                   // b*L
  float acc = conv_b[d];
#pragma unroll
  for (int j = 0; j < KC_; ++j) {
    int tt = t - (KC_ - 1) + j;
    if (tt >= 0)
      acc = fmaf(xz[(b0 + (size_t)tt) * (size_t)(2 * DI_) + d],
                 conv_w[d * KC_ + j], acc);
  }
  float s = acc * sigmoidf_(acc);   // SiLU
  u[i]  = s;
  ub[i] = (bf16_t)s;
}

// ---------------------------------------------------------------------------
// Selective scan. One lane per channel d (8192 channels), 16-state h in VGPRs;
// 2048 serial steps.  B_t/C_t reads are block-uniform -> scalar loads.
// __expf is a TRANS op -> co-executes with the VALU FMAs on CDNA5.
// ---------------------------------------------------------------------------
__global__ void __launch_bounds__(256)
scan_kernel(const float* __restrict__ delta, const float* __restrict__ u,
            const float* __restrict__ xdbl, const float* __restrict__ A_log,
            float* __restrict__ ys)
{
  const int d = blockIdx.x * 256 + threadIdx.x;   // channel
  const int b = blockIdx.y;
  float Arow[DS_], h[DS_];
#pragma unroll
  for (int n = 0; n < DS_; ++n) {
    Arow[n] = -__expf(A_log[d * DS_ + n]);        // A = -exp(A_log)
    h[n] = 0.0f;
  }
  const float* xrow = xdbl + (size_t)b * L_ * XD_;
  size_t base = ((size_t)b * L_) * DI_ + d;
  for (int t = 0; t < L_; ++t) {
    float dlt = delta[base + (size_t)t * DI_];
    float uv  = u[base + (size_t)t * DI_];
    float du  = dlt * uv;
    const float* xd = xrow + t * XD_;
    float y = 0.0f;
#pragma unroll
    for (int n = 0; n < DS_; ++n) {
      float Bt = xd[DR_ + n];                     // uniform across lanes
      float Ct = xd[DR_ + DS_ + n];
      h[n] = fmaf(__expf(dlt * Arow[n]), h[n], du * Bt);
      y    = fmaf(h[n], Ct, y);
    }
    ys[base + (size_t)t * DI_] = y;
  }
}

// ---------------------------------------------------------------------------
// yb = bf16( (ys + u*D) * SiLU(z) );  z = xz[:, :, DI:2*DI]
// ---------------------------------------------------------------------------
__global__ void __launch_bounds__(256)
gate_kernel(const float* __restrict__ xz, const float* __restrict__ u,
            const float* __restrict__ D_param, const float* __restrict__ ys,
            bf16_t* __restrict__ yb)
{
  size_t i   = (size_t)blockIdx.x * 256 + threadIdx.x;   // over B*L*DI
  int    d   = (int)(i % DI_);
  size_t row = i / DI_;
  float z = xz[row * (size_t)(2 * DI_) + DI_ + d];
  float g = z * sigmoidf_(z);
  yb[i] = (bf16_t)((ys[i] + u[i] * D_param[d]) * g);
}

// ---------------------------------------------------------------------------
extern "C" void kernel_launch(void* const* d_in, const int* in_sizes, int n_in,
                              void* d_out, int out_size, void* d_ws, size_t ws_size,
                              hipStream_t stream)
{
  const float* hidden     = (const float*)d_in[0];
  const float* in_proj_w  = (const float*)d_in[1];
  const float* conv_w     = (const float*)d_in[2];
  const float* conv_b     = (const float*)d_in[3];
  const float* x_proj_w   = (const float*)d_in[4];
  const float* dt_proj_w  = (const float*)d_in[5];
  const float* dt_proj_b  = (const float*)d_in[6];
  const float* A_log      = (const float*)d_in[7];
  const float* D_param    = (const float*)d_in[8];
  const float* out_proj_w = (const float*)d_in[9];
  float* out = (float*)d_out;

  const size_t ROWS = (size_t)B_ * L_;            // 8192

  // fp32 workspace
  float* ws    = (float*)d_ws;
  float* xz    = ws;                              // ROWS * 4096
  float* u     = xz    + ROWS * (size_t)(2 * DI_);
  float* xdbl  = u     + ROWS * (size_t)DI_;      // ROWS * 96
  float* delta = xdbl  + ROWS * (size_t)XD_;
  float* ys    = delta + ROWS * (size_t)DI_;
  // bf16 workspace (32B-aligned: all offsets multiples of 16 elements)
  bf16_t* bws   = (bf16_t*)(ys + ROWS * (size_t)DI_);
  bf16_t* hid_b = bws;                                   // ROWS * DM
  bf16_t* win_b = hid_b + ROWS * (size_t)DM_;            // 2*DI * DM
  bf16_t* wx_b  = win_b + (size_t)(2 * DI_) * DM_;       // XD * DI
  bf16_t* wdt_b = wx_b  + (size_t)XD_ * DI_;             // DI * DR
  bf16_t* wo_b  = wdt_b + (size_t)DI_ * DR_;             // DM * DI
  bf16_t* u_b   = wo_b  + (size_t)DM_ * DI_;             // ROWS * DI
  bf16_t* dt_b  = u_b   + ROWS * (size_t)DI_;            // ROWS * DR
  bf16_t* y_b   = dt_b  + ROWS * (size_t)DR_;            // ROWS * DI

  // 0) one-pass f32 -> bf16 of GEMM operands (amortized 6..512x per reuse)
  cvt_f32_bf16<<<dim3((unsigned)(ROWS * DM_ / 1024)), 256, 0, stream>>>(hidden, hid_b);
  cvt_f32_bf16<<<dim3((unsigned)((size_t)2 * DI_ * DM_ / 1024)), 256, 0, stream>>>(in_proj_w, win_b);
  cvt_f32_bf16<<<dim3((unsigned)((size_t)XD_ * DI_ / 1024)), 256, 0, stream>>>(x_proj_w, wx_b);
  cvt_f32_bf16<<<dim3((unsigned)((size_t)DI_ * DR_ / 1024)), 256, 0, stream>>>(dt_proj_w, wdt_b);
  cvt_f32_bf16<<<dim3((unsigned)((size_t)DM_ * DI_ / 1024)), 256, 0, stream>>>(out_proj_w, wo_b);

  // 1) in_proj: xz = hidden @ in_proj_w^T     (M=8192, N=4096, K=1024)
  {
    dim3 grid((2 * DI_) / 512, ROWS / 16);
    wmma_nt_gemm<0, 4><<<grid, 256, 0, stream>>>(hid_b, DM_, win_b, DM_,
                                                 xz, 2 * DI_, 2 * DI_, DM_, nullptr);
  }
  // 2) causal depthwise conv + SiLU -> u (f32 + bf16)
  {
    size_t total = ROWS * (size_t)DI_;
    conv_silu_kernel<<<dim3((unsigned)(total / 256)), 256, 0, stream>>>(
        xz, conv_w, conv_b, u, u_b);
  }
  // 3) x_proj: xdbl = u @ x_proj_w^T          (M=8192, N=96, K=2048)
  {
    dim3 grid(1, ROWS / 16);                     // 6 valid 16-tiles of N=96
    wmma_nt_gemm<0, 1><<<grid, 256, 0, stream>>>(u_b, DI_, wx_b, DI_,
                                                 xdbl, XD_, XD_, DI_, nullptr);
  }
  // 3b) pack dt columns (xdbl[:, 0:64]) to bf16, lda=64
  extract_dt_bf16<<<dim3((unsigned)(ROWS * 16 / 256)), 256, 0, stream>>>(xdbl, dt_b);

  // 4) delta = softplus(dt @ dt_proj_w^T + b) (M=8192, N=2048, K=64)
  {
    dim3 grid(DI_ / 512, ROWS / 16);
    wmma_nt_gemm<1, 4><<<grid, 256, 0, stream>>>(dt_b, DR_, wdt_b, DR_,
                                                 delta, DI_, DI_, DR_, dt_proj_b);
  }
  // 5) selective scan -> ys (f32 recurrence)
  {
    dim3 grid(DI_ / 256, B_);
    scan_kernel<<<grid, 256, 0, stream>>>(delta, u, xdbl, A_log, ys);
  }
  // 6) gate: y_b = bf16((ys + u*D) * SiLU(z))
  {
    size_t total = ROWS * (size_t)DI_;
    gate_kernel<<<dim3((unsigned)(total / 256)), 256, 0, stream>>>(
        xz, u, D_param, ys, y_b);
  }
  // 7) out_proj: out = y @ out_proj_w^T       (M=8192, N=1024, K=2048)
  {
    dim3 grid(DM_ / 512, ROWS / 16);
    wmma_nt_gemm<0, 4><<<grid, 256, 0, stream>>>(y_b, DI_, wo_b, DI_,
                                                 out, DM_, DM_, DI_, nullptr);
  }
}
